// HardLSTM_8340826489640
// MI455X (gfx1250) — matrix operand order repaired and checked by reading, not compiled, not run on hardware
//
#include <hip/hip_runtime.h>

// ---------------------------------------------------------------------------
// HardLSTM (2 layers, T=512, B=32, IN=H=1024) for MI455X / gfx1250.
//
// Strategy:
//  * bf16 WMMA (v_wmma_f32_16x16x32_bf16) for all GEMM work; weights converted
//    once to bf16 (32 MB total -> resident in 192 MB L2 across all steps).
//  * Per recurrent step: one fused GEMM  gates[32,4096] = A[32,2048] x W^T
//    where A = [x_t | h] concatenated along K.
//  * eps = mean(x_t) handled as a rank-1 correction via precomputed
//    per-gate-row weight sums:  (x+eps)W^T = xW^T + eps*rowsum(W).
//  * Each wave owns one 16-unit h-tile and computes all 4 gate tiles
//    (4 independent WMMA accumulator chains), so the LSTM cell elementwise
//    update is wave-local.  A-panel staged in LDS (XOR-swizzled) per WG.
// ---------------------------------------------------------------------------

#define T_STEPS 512
#define BB      32
#define INSZ    1024
#define HSZ     1024
#define KTOT    2048          // IN + H fused K dimension
#define NG      4096          // 4*H gate rows

typedef __attribute__((ext_vector_type(16))) __bf16 v16bf;
typedef __attribute__((ext_vector_type(8)))  float  v8f;

union ABCast { uint4 u[2]; v16bf v; };

__device__ __forceinline__ unsigned short f2bf(float f) {
  unsigned int u = __float_as_uint(f);                 // round-to-nearest-even
  return (unsigned short)((u + 0x7FFFu + ((u >> 16) & 1u)) >> 16);
}
__device__ __forceinline__ float hsig(float x)  { return fminf(fmaxf(0.2f * x + 0.5f, 0.0f), 1.0f); }
__device__ __forceinline__ float htanh(float x) { return fminf(fmaxf(x, -1.0f), 1.0f); }

// chunk = 8 ushorts (16B).  XOR-swizzle chunk index by row so the 16 rows read
// by one A-fragment fetch hit different bank groups (row stride 4096B aliases
// all 64 banks otherwise).
__device__ __forceinline__ int lds_chunk(int row, int chunk) {
  return row * (KTOT / 8) + (chunk ^ (row & 7));
}

// ---------------------------------------------------------------------------
// Prep: concat [W_ih | W_hh] row-major as bf16, rowsums (for eps rank-1 term),
// folded bias bc[n] = b_ih[n] + b_hh[n].
// ---------------------------------------------------------------------------
__global__ __launch_bounds__(256)
void prep_weights(const float* __restrict__ Wih, const float* __restrict__ Whh,
                  const float* __restrict__ bih, const float* __restrict__ bhh,
                  unsigned short* __restrict__ Wcat,
                  float* __restrict__ rs, float* __restrict__ bc) {
  const int n = blockIdx.x;                 // gate row 0..4095
  const float* a = Wih + (size_t)n * INSZ;
  const float* b = Whh + (size_t)n * HSZ;
  unsigned short* w = Wcat + (size_t)n * KTOT;
  float s = 0.0f;
  for (int k = threadIdx.x; k < INSZ; k += 256) { float v = a[k]; s += v; w[k] = f2bf(v); }
  for (int k = threadIdx.x; k < HSZ;  k += 256) { float v = b[k]; s += v; w[INSZ + k] = f2bf(v); }
  __shared__ float red[256];
  red[threadIdx.x] = s;
  __syncthreads();
  for (int off = 128; off > 0; off >>= 1) {
    if (threadIdx.x < off) red[threadIdx.x] += red[threadIdx.x + off];
    __syncthreads();
  }
  if (threadIdx.x == 0) { rs[n] = red[0]; bc[n] = bih[n] + bhh[n]; }
}

// ---------------------------------------------------------------------------
// Prep: layer-0 input Xi = bf16(x + 1), eps0[t] = mean(x_t + 1); zero eps1.
// ---------------------------------------------------------------------------
__global__ __launch_bounds__(256)
void prep_x(const float* __restrict__ x, unsigned short* __restrict__ Xi,
            float* __restrict__ eps0, float* __restrict__ eps1) {
  const int t = blockIdx.x;
  const float* xt = x + (size_t)t * BB * INSZ;
  unsigned short* xo = Xi + (size_t)t * BB * INSZ;
  float s = 0.0f;
  for (int i = threadIdx.x; i < BB * INSZ; i += 256) {
    float v = xt[i] + 1.0f;                 // StackedLSTM adds `one`
    s += v;
    xo[i] = f2bf(v);
  }
  __shared__ float red[256];
  red[threadIdx.x] = s;
  __syncthreads();
  for (int off = 128; off > 0; off >>= 1) {
    if (threadIdx.x < off) red[threadIdx.x] += red[threadIdx.x + off];
    __syncthreads();
  }
  if (threadIdx.x == 0) {
    eps0[t] = red[0] * (1.0f / (BB * INSZ));
    eps1[t] = 0.0f;                         // accumulated by layer-0 epilogue
  }
}

// ---------------------------------------------------------------------------
// Prep: initial h (bf16) and c (f32 working copy) for both layers.
// ---------------------------------------------------------------------------
__global__ __launch_bounds__(256)
void prep_state(const float* __restrict__ h0, const float* __restrict__ c0,
                unsigned short* __restrict__ hbf, float* __restrict__ cbuf) {
  const int i = blockIdx.x * 256 + threadIdx.x;   // 2*B*H total
  hbf[i]  = f2bf(h0[i]);
  cbuf[i] = c0[i];
}

// ---------------------------------------------------------------------------
// One recurrent step.  grid = 16 WGs (mtile in {0,1} x 8 h-groups),
// block = 256 (8 waves).  Wave owns h-tile = hgroup*8 + wave (16 h units) and
// computes all four 16x16 gate tiles over K=2048 (64 WMMAs per gate).
// ---------------------------------------------------------------------------
template <int LAYER>
__global__ __launch_bounds__(256)
void lstm_step(const unsigned short* __restrict__ Xbf,   // layer input bf16 [T,B,1024]
               const unsigned short* __restrict__ Wcat,  // [4096,2048] bf16
               const float* __restrict__ rs,             // [4096] rowsums
               const float* __restrict__ bc,             // [4096] folded bias
               const float* __restrict__ eps_arr,        // [T] eps for this layer
               float* __restrict__ eps_next,             // layer0: eps1 accum
               unsigned short* __restrict__ hbf,         // [B,H] recurrent h (bf16)
               float* __restrict__ cst,                  // [B,H] cell state (f32)
               unsigned short* __restrict__ ybf,         // layer0 output (bf16)
               float* __restrict__ yf,                   // layer1 output (f32, d_out)
               float* __restrict__ hn_out,               // [B,H] final h slab
               float* __restrict__ cn_out,               // [B,H] final c slab
               int t) {
  __shared__ unsigned short Abuf[16 * KTOT];             // 64 KB A-panel

  const int tid   = threadIdx.x;
  const int mtile = blockIdx.x & 1;
  const int hgrp  = blockIdx.x >> 1;
  const int wave  = tid >> 5;
  const int lane  = tid & 31;
  const int htile = hgrp * 8 + wave;

  // ---- stage A = [x_t | h] (16 rows x 2048 bf16) into LDS (swizzled) ----
  for (int c = tid; c < 16 * (KTOT / 8); c += 256) {
    const int row = c >> 8;                  // 256 chunks per row
    const int ck  = c & 255;
    const int kc  = ck * 8;
    const int b   = mtile * 16 + row;
    uint4 v;
    if (kc < INSZ) {
      v = *reinterpret_cast<const uint4*>(Xbf + ((size_t)t * BB + b) * INSZ + kc);
    } else {
      v = *reinterpret_cast<const uint4*>(hbf + (size_t)b * HSZ + (kc - INSZ));
    }
    *reinterpret_cast<uint4*>(&Abuf[lds_chunk(row, ck) * 8]) = v;
  }
  __syncthreads();

  const float eps = eps_arr[t];

  v8f acc0 = {}, acc1 = {}, acc2 = {}, acc3 = {};

  const int arow  = lane & 15;               // A row handled by this lane
  const int ahalf = lane >> 4;               // which K-half of the fragment
  const int ncol  = lane & 15;               // B/D column handled by this lane
  const int hidx  = htile * 16 + ncol;

  const unsigned short* w0 = Wcat + (size_t)(0 * HSZ + hidx) * KTOT;
  const unsigned short* w1 = Wcat + (size_t)(1 * HSZ + hidx) * KTOT;
  const unsigned short* w2 = Wcat + (size_t)(2 * HSZ + hidx) * KTOT;
  const unsigned short* w3 = Wcat + (size_t)(3 * HSZ + hidx) * KTOT;
  const int bko = ahalf * 8;                 // B frag: K = [ahalf*8, +8) and [16+ahalf*8, +8)

  for (int k = 0; k < KTOT; k += 32) {
    // A fragment: lane holds row arow, K = [k+ahalf*8, +8) and [k+16+ahalf*8, +8)
    ABCast a;
    const int c0i = (k >> 3) + ahalf;
    a.u[0] = *reinterpret_cast<const uint4*>(&Abuf[lds_chunk(arow, c0i) * 8]);
    a.u[1] = *reinterpret_cast<const uint4*>(&Abuf[lds_chunk(arow, c0i + 2) * 8]);

    const int ko = k + bko;
    ABCast b0, b1, b2, b3;
    b0.u[0] = *reinterpret_cast<const uint4*>(w0 + ko);
    b0.u[1] = *reinterpret_cast<const uint4*>(w0 + ko + 16);
    b1.u[0] = *reinterpret_cast<const uint4*>(w1 + ko);
    b1.u[1] = *reinterpret_cast<const uint4*>(w1 + ko + 16);
    b2.u[0] = *reinterpret_cast<const uint4*>(w2 + ko);
    b2.u[1] = *reinterpret_cast<const uint4*>(w2 + ko + 16);
    b3.u[0] = *reinterpret_cast<const uint4*>(w3 + ko);
    b3.u[1] = *reinterpret_cast<const uint4*>(w3 + ko + 16);

    __builtin_prefetch(w0 + ko + 32, 0, 1);  // global_prefetch_b8: next K tile

    acc0 = __builtin_amdgcn_wmma_f32_16x16x32_bf16(false, a.v, false, b0.v, (short)0, acc0, false, false);
    acc1 = __builtin_amdgcn_wmma_f32_16x16x32_bf16(false, a.v, false, b1.v, (short)0, acc1, false, false);
    acc2 = __builtin_amdgcn_wmma_f32_16x16x32_bf16(false, a.v, false, b2.v, (short)0, acc2, false, false);
    acc3 = __builtin_amdgcn_wmma_f32_16x16x32_bf16(false, a.v, false, b3.v, (short)0, acc3, false, false);
  }

  // ---- LSTM cell epilogue (wave-local: all 4 gates for this 16x16 patch) ----
  const float bci = bc[hidx]           + eps * rs[hidx];
  const float bcf = bc[HSZ + hidx]     + eps * rs[HSZ + hidx];
  const float bcg = bc[2 * HSZ + hidx] + eps * rs[2 * HSZ + hidx];
  const float bco = bc[3 * HSZ + hidx] + eps * rs[3 * HSZ + hidx];

  float bsum = 0.0f;
#pragma unroll
  for (int r = 0; r < 8; ++r) {
    const int b = mtile * 16 + ahalf * 8 + r;        // C layout: M = r + 8*(lane>=16)
    const float ig = hsig(acc0[r] + bci);
    const float fg = hsig(acc1[r] + bcf);
    const float gv = htanh(acc2[r] + bcg);
    const float og = hsig(acc3[r] + bco);
    const size_t sidx = (size_t)b * HSZ + hidx;
    const float cc = cst[sidx] + eps;
    const float cy = fg * cc + ig * gv;
    const float hy = og * htanh(cy);
    cst[sidx] = cy;
    hbf[sidx] = f2bf(hy);
    if (LAYER == 0) {
      ybf[((size_t)t * BB + b) * HSZ + hidx] = f2bf(hy);
      bsum += hy;
    } else {
      yf[((size_t)t * BB + b) * HSZ + hidx] = hy;
    }
    if (t == T_STEPS - 1) { hn_out[sidx] = hy; cn_out[sidx] = cy; }
  }
  if (LAYER == 0) {
    atomicAdd(eps_next + t, bsum * (1.0f / (BB * HSZ)));  // eps for layer 1
  }
}

// ---------------------------------------------------------------------------
extern "C" void kernel_launch(void* const* d_in, const int* in_sizes, int n_in,
                              void* d_out, int out_size, void* d_ws, size_t ws_size,
                              hipStream_t stream) {
  (void)in_sizes; (void)n_in; (void)out_size; (void)ws_size;

  const float* x    = (const float*)d_in[0];
  const float* h0   = (const float*)d_in[1];
  const float* c0   = (const float*)d_in[2];
  const float* Wih0 = (const float*)d_in[3];
  const float* Whh0 = (const float*)d_in[4];
  const float* bih0 = (const float*)d_in[5];
  const float* bhh0 = (const float*)d_in[6];
  const float* Wih1 = (const float*)d_in[7];
  const float* Whh1 = (const float*)d_in[8];
  const float* bih1 = (const float*)d_in[9];
  const float* bhh1 = (const float*)d_in[10];

  char* p = (char*)d_ws;
  auto take = [&](size_t bytes) -> char* {
    char* r = p;
    p += (bytes + 255) & ~(size_t)255;
    return r;
  };
  unsigned short* Wcat0 = (unsigned short*)take((size_t)NG * KTOT * 2);   // 16 MB
  unsigned short* Wcat1 = (unsigned short*)take((size_t)NG * KTOT * 2);   // 16 MB
  unsigned short* Xi0   = (unsigned short*)take((size_t)T_STEPS * BB * INSZ * 2); // 32 MB
  unsigned short* Y0    = (unsigned short*)take((size_t)T_STEPS * BB * HSZ * 2);  // 32 MB
  float* rs0  = (float*)take(NG * 4);
  float* bc0  = (float*)take(NG * 4);
  float* rs1  = (float*)take(NG * 4);
  float* bc1  = (float*)take(NG * 4);
  float* eps0 = (float*)take(T_STEPS * 4);
  float* eps1 = (float*)take(T_STEPS * 4);
  unsigned short* hbf = (unsigned short*)take((size_t)2 * BB * HSZ * 2);
  float* cbuf = (float*)take((size_t)2 * BB * HSZ * 4);

  float* yout = (float*)d_out;                               // [T,B,H]
  float* hn   = yout + (size_t)T_STEPS * BB * HSZ;           // [2,B,H]
  float* cn   = hn + (size_t)2 * BB * HSZ;                   // [2,B,H]

  prep_weights<<<NG, 256, 0, stream>>>(Wih0, Whh0, bih0, bhh0, Wcat0, rs0, bc0);
  prep_weights<<<NG, 256, 0, stream>>>(Wih1, Whh1, bih1, bhh1, Wcat1, rs1, bc1);
  prep_x<<<T_STEPS, 256, 0, stream>>>(x, Xi0, eps0, eps1);
  prep_state<<<(2 * BB * HSZ) / 256, 256, 0, stream>>>(h0, c0, hbf, cbuf);

  for (int t = 0; t < T_STEPS; ++t) {
    lstm_step<0><<<16, 256, 0, stream>>>(Xi0, Wcat0, rs0, bc0, eps0, eps1,
                                         hbf, cbuf, Y0, nullptr, hn, cn, t);
  }
  for (int t = 0; t < T_STEPS; ++t) {
    lstm_step<1><<<16, 256, 0, stream>>>(Y0, Wcat1, rs1, bc1, eps1, nullptr,
                                         hbf + (size_t)BB * HSZ, cbuf + (size_t)BB * HSZ,
                                         nullptr, yout,
                                         hn + (size_t)BB * HSZ, cn + (size_t)BB * HSZ, t);
  }
}
